// ARFSA_51994874085964
// MI455X (gfx1250) — matrix operand order repaired and checked by reading, not compiled, not run on hardware
//
#include <hip/hip_runtime.h>
#include <hip/hip_bf16.h>

// ---- problem constants (from reference) ----
#define BATCH 32
#define NCH   64      // IN_C == OUT_C == 64
#define S     128
#define HW    (S * S) // 16384

typedef _Float16 half_t;
typedef __attribute__((ext_vector_type(16))) _Float16 v16h;
typedef __attribute__((ext_vector_type(8)))  float    v8f;
typedef __attribute__((ext_vector_type(4)))  unsigned int u32x4;

union FragH { v16h h; u32x4 u[2]; };

// CDNA5 async copy: global -> LDS direct (no VGPR round trip), ASYNCcnt-tracked.
__device__ __forceinline__ void async_g2lds_b128(unsigned lds_off, const void* gaddr) {
    asm volatile("global_load_async_to_lds_b128 %0, %1, off"
                 :: "v"(lds_off), "v"(gaddr) : "memory");
}
__device__ __forceinline__ void wait_async0() {
    asm volatile("s_wait_asynccnt 0" ::: "memory");
}
// flat shared address -> LDS byte offset (low 32 bits of the address)
__device__ __forceinline__ unsigned lds_off(const void* p) {
    return (unsigned)(size_t)p;
}

// =====================================================================
// Kernel 1: q/k/v projections (64x64 channel-mix GEMM), pc folded into k.
//   q, k' -> f16 workspace ; v -> f32 directly into d_out (staging).
// grid: BATCH * (HW/128) blocks of 256 threads (8 waves x 16 pixels).
// =====================================================================
__global__ __launch_bounds__(256)
void proj_qkv_kernel(const float* __restrict__ x,
                     const float* __restrict__ wq, const float* __restrict__ bq,
                     const float* __restrict__ wk, const float* __restrict__ bk,
                     const float* __restrict__ wv, const float* __restrict__ bv,
                     const float* __restrict__ pos_code,
                     half_t* __restrict__ qh, half_t* __restrict__ kh,
                     float* __restrict__ vstage)
{
    __shared__ __attribute__((aligned(16))) half_t Wlds[3][NCH * NCH]; // f16 weights
    __shared__ __attribute__((aligned(16))) float  blds[3][NCH];
    __shared__ __attribute__((aligned(16))) half_t xT[128 * NCH];      // [pixel][channel]

    const int t  = threadIdx.x;
    const int b  = blockIdx.x >> 7;           // 128 pixel-tiles per batch
    const int pb = (blockIdx.x & 127) * 128;  // pixel base of this tile

    // stage weights (f16) + biases  (needs f32->f16 convert, so via VGPRs)
    for (int i = t; i < NCH * NCH; i += 256) {
        Wlds[0][i] = (half_t)wq[i];
        Wlds[1][i] = (half_t)wk[i];
        Wlds[2][i] = (half_t)wv[i];
    }
    if (t < NCH) { blds[0][t] = bq[t]; blds[1][t] = bk[t]; blds[2][t] = bv[t]; }

    // stage x tile transposed: xT[pixel][channel], f16 (coalesced global reads)
    const float* xb = x + (size_t)b * NCH * HW;
    for (int i = t; i < 128 * NCH; i += 256) {
        const int c  = i >> 7;
        const int pl = i & 127;
        xT[pl * NCH + c] = (half_t)xb[(size_t)c * HW + pb + pl];
    }
    __syncthreads();

    const int lane = t & 31;
    const int wid  = t >> 5;
    const int m    = lane & 15;
    const int hi   = lane >> 4;
    const int npix = wid * 16 + m;   // B-fragment column (local pixel)
    const int p    = pb + npix;      // global flat pixel

    // B fragments (x), one per K-step (K=64 -> 2 steps of 32).
    // B layout: lane n holds K = 16*hi + 0..15 contiguous -> two b128 loads.
    FragH bf[2];
#pragma unroll
    for (int ks = 0; ks < 2; ++ks) {
        const half_t* src = &xT[npix * NCH + ks * 32 + hi * 16];
        bf[ks].u[0] = *(const u32x4*)(src);
        bf[ks].u[1] = *(const u32x4*)(src + 8);
    }

    const float pcv = pos_code[p];   // pc identical across channels -> channel 0

#pragma unroll
    for (int wsel = 0; wsel < 3; ++wsel) {
        v8f acc[4] = {};             // 4 M-tiles of 16 out-channels
#pragma unroll
        for (int ks = 0; ks < 2; ++ks) {
#pragma unroll
            for (int mt = 0; mt < 4; ++mt) {
                // A layout: lane m row, K = 16*hi + 0..15 contiguous (+32*ks)
                const int o = mt * 16 + m;
                const half_t* ws = &Wlds[wsel][o * NCH + ks * 32 + hi * 16];
                FragH af;
                af.u[0] = *(const u32x4*)(ws);
                af.u[1] = *(const u32x4*)(ws + 8);
                acc[mt] = __builtin_amdgcn_wmma_f32_16x16x32_f16(
                    false, af.h, false, bf[ks].h, (short)0, acc[mt], false, false);
            }
        }
        // epilogue: C layout -> row M = r + 8*hi, col N = m (pixel npix)
#pragma unroll
        for (int mt = 0; mt < 4; ++mt) {
#pragma unroll
            for (int r = 0; r < 8; ++r) {
                const int o = mt * 16 + r + hi * 8;
                const float val = acc[mt][r] + blds[wsel][o];
                const size_t idx = ((size_t)b * NCH + o) * HW + p;
                if      (wsel == 0) qh[idx] = (half_t)val;
                else if (wsel == 1) kh[idx] = (half_t)(val + pcv); // k' = k + pc
                else                vstage[idx] = val;             // v (f32) staged in d_out
            }
        }
    }
}

// =====================================================================
// Kernel 2: per (b,c): att = q @ k'^T (128x128x128), global softmax over
// all 16384 logits, out = softmax(att) * v.  One workgroup per (b,c).
// q/k' tiles staged with CDNA5 async global->LDS copies (ASYNCcnt path).
// =====================================================================
__global__ __launch_bounds__(256)
void attn_softmax_kernel(const half_t* __restrict__ qh,
                         const half_t* __restrict__ kh,
                         float* out)   // read v (staged), write att*v
{
    __shared__ __attribute__((aligned(16))) half_t qs[HW]; // 32 KB
    __shared__ __attribute__((aligned(16))) half_t ks[HW]; // 32 KB

    const int t  = threadIdx.x;
    const int bc = blockIdx.x;                 // 0 .. BATCH*NCH-1
    const size_t base = (size_t)bc * HW;

    // stage q and k' tiles in LDS via async b128 copies (no VGPR round trip)
    {
        const u32x4* qg = (const u32x4*)(qh + base);
        const u32x4* kg = (const u32x4*)(kh + base);
        const unsigned ql = lds_off(qs);
        const unsigned kl = lds_off(ks);
        for (int i = t; i < HW / 8; i += 256) {
            async_g2lds_b128(ql + i * 16, qg + i);
            async_g2lds_b128(kl + i * 16, kg + i);
        }
        wait_async0();
    }
    __syncthreads();

    const int lane = t & 31;
    const int wid  = t >> 5;
    const int m    = lane & 15;
    const int hi   = lane >> 4;
    const int row  = wid * 16 + m;             // A-fragment row (q row h)

    v8f acc[8] = {};                           // 16 rows x 128 cols of logits
#pragma unroll
    for (int kstep = 0; kstep < 4; ++kstep) {  // K = 128 -> 4 steps of 32
        FragH af;
        const half_t* asrc = &qs[row * S + kstep * 32 + hi * 16];
        af.u[0] = *(const u32x4*)(asrc);
        af.u[1] = *(const u32x4*)(asrc + 8);
#pragma unroll
        for (int nt = 0; nt < 8; ++nt) {
            // B[k][n] = k'[n, k]: lane n = nt*16+m reads 16 contiguous w's
            const half_t* bsrc = &ks[(nt * 16 + m) * S + kstep * 32 + hi * 16];
            FragH bf;
            bf.u[0] = *(const u32x4*)(bsrc);
            bf.u[1] = *(const u32x4*)(bsrc + 8);
            acc[nt] = __builtin_amdgcn_wmma_f32_16x16x32_f16(
                false, af.h, false, bf.h, (short)0, acc[nt], false, false);
        }
    }

    // ---- global softmax over the whole 128x128 slab ----
    __syncthreads();                       // q/k LDS no longer needed
    float* red = (float*)qs;               // alias 32B of reduction scratch

    float mx = -3.402823e38f;
#pragma unroll
    for (int nt = 0; nt < 8; ++nt)
#pragma unroll
        for (int r = 0; r < 8; ++r) mx = fmaxf(mx, acc[nt][r]);
#pragma unroll
    for (int off = 16; off > 0; off >>= 1) mx = fmaxf(mx, __shfl_xor(mx, off, 32));
    if (lane == 0) red[wid] = mx;
    __syncthreads();
    float gmx = red[0];
#pragma unroll
    for (int w = 1; w < 8; ++w) gmx = fmaxf(gmx, red[w]);

    float sum = 0.f;
#pragma unroll
    for (int nt = 0; nt < 8; ++nt)
#pragma unroll
        for (int r = 0; r < 8; ++r) {
            const float e = __expf(acc[nt][r] - gmx);
            acc[nt][r] = e;
            sum += e;
        }
#pragma unroll
    for (int off = 16; off > 0; off >>= 1) sum += __shfl_xor(sum, off, 32);
    __syncthreads();                       // everyone done reading maxes
    if (lane == 0) red[wid] = sum;
    __syncthreads();
    float tot = 0.f;
#pragma unroll
    for (int w = 0; w < 8; ++w) tot += red[w];
    const float inv = 1.0f / tot;

    // ---- out = softmax * v (v staged in-place in `out`) ----
    float* ob = out + base;
#pragma unroll
    for (int nt = 0; nt < 8; ++nt) {
#pragma unroll
        for (int r = 0; r < 8; ++r) {
            const int h   = wid * 16 + r + hi * 8;   // C layout: M = r + 8*hi
            const int col = nt * 16 + m;             // N = lane 0..15
            const size_t idx = (size_t)h * S + col;
            const float v = ob[idx];                 // read-before-write, same lane
            ob[idx] = acc[nt][r] * inv * v;
        }
    }
}

// =====================================================================
extern "C" void kernel_launch(void* const* d_in, const int* in_sizes, int n_in,
                              void* d_out, int out_size, void* d_ws, size_t ws_size,
                              hipStream_t stream) {
    const float* x   = (const float*)d_in[0];
    const float* wq  = (const float*)d_in[1];
    const float* bq  = (const float*)d_in[2];
    const float* wk  = (const float*)d_in[3];
    const float* bk  = (const float*)d_in[4];
    const float* wv  = (const float*)d_in[5];
    const float* bv  = (const float*)d_in[6];
    const float* pos = (const float*)d_in[7];

    float* out = (float*)d_out;

    const size_t nElem = (size_t)BATCH * NCH * HW;      // 33,554,432
    half_t* qh = (half_t*)d_ws;                         // 64 MB
    half_t* kh = qh + nElem;                            // 64 MB

    // Kernel 1: projections (q,k'+pc -> f16 ws; v -> f32 staged in d_out)
    proj_qkv_kernel<<<dim3(BATCH * (HW / 128)), dim3(256), 0, stream>>>(
        x, wq, bq, wk, bk, wv, bv, pos, qh, kh, out);

    // Kernel 2: fused attention matmul + global softmax + *v
    attn_softmax_kernel<<<dim3(BATCH * NCH), dim3(256), 0, stream>>>(qh, kh, out);
}
